// RelFeatureExtractor_37323265803095
// MI455X (gfx1250) — matrix operand order repaired and checked by reading, not compiled, not run on hardware
//
#include <hip/hip_runtime.h>

// ---------------- types ----------------
typedef __attribute__((ext_vector_type(16))) __bf16       v16bf;
typedef __attribute__((ext_vector_type(8)))  float        v8f;
typedef __attribute__((ext_vector_type(4)))  unsigned int u32x4;
typedef __attribute__((ext_vector_type(4)))  float        f32x4;
typedef unsigned short u16b;

#define NB    1024   // batch
#define TT    128    // seq len
#define HH    512    // hidden
#define DD    520    // H + EV
#define DP    544    // D padded to 17*32
#define CC    100    // conv channels
#define CP    112    // C padded to 7*16
#define FEATN 2052
#define FEATP 2080   // 65*32
#define REX   136    // 4 zero rows + 128 data rows + 4 zero rows

union FragBF { u32x4 q[2]; u16b s[16]; v16bf v; };
union Pack8  { u32x4 q;    u16b s[8]; };

__device__ __forceinline__ u16b f2bf(float x) {
  unsigned u = __float_as_uint(x);
  return (u16b)((u + 0x7FFFu + ((u >> 16) & 1u)) >> 16);  // RNE
}

__device__ __forceinline__ v8f wmma_bf16(const FragBF& a, const FragBF& b, v8f c) {
  return __builtin_amdgcn_wmma_f32_16x16x32_bf16(false, a.v, false, b.v, (short)0, c,
                                                 false, false);
}

struct ConvW { const float* w[9]; const float* b[9]; };   // [branch*3 + ki]

// -------- prep: reorder conv weights -> bf16 [slab][c][d], pack biases --------
__global__ __launch_bounds__(256) void prep_conv_w_kernel(ConvW cw, u16b* __restrict__ Wr,
                                                          float* __restrict__ biasbuf) {
  int gid = blockIdx.x * 256 + threadIdx.x;
  if (gid < 9 * CP) {
    int slot = gid / CP, c = gid - slot * CP;
    biasbuf[gid] = (c < CC) ? cw.b[slot][c] : 0.f;
  }
  const int tot = 36 * CP * DP;                    // 36 slabs = 3 branches * (3+4+5) taps
  for (int i = gid; i < tot; i += (int)gridDim.x * 256) {
    int slab = i / (CP * DP);
    int r = i - slab * (CP * DP);
    int c = r / DP, d = r - c * DP;
    int br = slab / 12, s2 = slab - br * 12;
    int ki, tap;
    if (s2 < 3)      { ki = 0; tap = s2; }
    else if (s2 < 7) { ki = 1; tap = s2 - 3; }
    else             { ki = 2; tap = s2 - 7; }
    int K = 3 + ki;
    float v = 0.f;
    if (c < CC && d < DD) v = cw.w[br * 3 + ki][(c * DD + d) * K + tap];
    Wr[i] = f2bf(v);
  }
}

// -------- prep: transpose MLP weights -> bf16 [h][f], zero-padded K --------
__global__ __launch_bounds__(256) void prep_mlp_w_kernel(const float* __restrict__ mw,
                                                         u16b* __restrict__ Wm) {
  const int tot = HH * FEATP;
  for (int i = blockIdx.x * 256 + threadIdx.x; i < tot; i += (int)gridDim.x * 256) {
    int h = i / FEATP, f = i - h * FEATP;
    float v = (f < FEATN) ? mw[f * HH + h] : 0.f;
    Wm[i] = f2bf(v);
  }
}

// -------- zero feat rows + fill L / R / dist features --------
__global__ __launch_bounds__(256) void fill_feat_kernel(const float* __restrict__ seq,
    const int* __restrict__ span, const int* __restrict__ slens, float* __restrict__ feat) {
  int b = blockIdx.x, tid = threadIdx.x;
  float* fr = feat + (size_t)b * FEATP;
  for (int f = tid; f < FEATP; f += 256) fr[f] = 0.f;
  __syncthreads();
  int e1s = span[b * 4 + 0], e1e = span[b * 4 + 1];
  int e2s = span[b * 4 + 2], e2e = span[b * 4 + 3];
  int slen = slens[b];
  const float* sb = seq + (size_t)b * TT * HH;
  for (int h = tid; h < HH; h += 256) {
    float lv = 0.f;                                  // L = bilstm_minus(seq, 0, e1s)
    if (e1s > 0) lv = sb[(size_t)min(max(e1s - 1, 0), TT - 1) * HH + h];
    fr[900 + h] = lv;
    float rv = 0.f;                                  // R = bilstm_minus(seq, e2e, slen)
    if (e2e < slen) {
      float ve = sb[(size_t)(TT - 1) * HH + h];
      rv = (e2e == 0) ? ve : ve - sb[(size_t)min(max(e2e - 1, 0), TT - 1) * HH + h];
    }
    fr[1412 + h] = rv;
  }
  if (tid == 0) {
    int dd = e2s - e1e;
    if (dd >= 0 && dd < TT) fr[1924 + dd] = 1.f;     // one-hot distance
  }
}

// -------- per-tile epilogue: bias + relu + masked time-max + LDS max --------
__device__ __forceinline__ void reduce_store(const v8f& acc, int mt, int c, int g,
                                             int rowsValid, const float* __restrict__ biasrow,
                                             unsigned* __restrict__ mxrow) {
  float bias = biasrow[c];
  float m = 0.f;
  #pragma unroll
  for (int r = 0; r < 8; ++r) {
    int orow = mt * 16 + g * 8 + r;
    float y = acc[r] + bias;
    y = (y > 0.f) ? y : 0.f;
    if (orow < rowsValid) m = fmaxf(m, y);           // per-lane select, no EXEC change
  }
  m = fmaxf(m, __shfl_xor(m, 16, 32));               // combine row halves
  atomicMax(&mxrow[c], __float_as_uint(m));          // monotone for floats >= 0
}

// -------- one 2x2 block of 16x16 tiles; K, edge-validity compile-time --------
template<int K, bool M1, bool N1>
__device__ __forceinline__ void conv_block(
    const u16b* __restrict__ Xe, const u16b* __restrict__ Wb,
    unsigned* __restrict__ mxrow, const float* __restrict__ biasrow,
    int tr0, int c0, int g, int mt0) {
  constexpr int itTot = 17 * K;                      // (kc, tap) pairs, tap fastest
  constexpr int rowsValid = TT + K - 1;
  constexpr int slabE = CP * DP;                     // elements per tap slab
  v8f acc00 = {}, acc01 = {}, acc10 = {}, acc11 = {};
  int lkc = 0, ltap = 0;                             // next (kc, tap) to load (uniform)
  // Hoisted bases; deltas (ltap*DP, lkc*32, +16*DP, +16, +8) fold into offsets.
  const u16b* Xb = Xe + (tr0 + (5 - K)) * DP + g * 8;   // in-range whenever M1
  const u16b* Wp = Wb + (size_t)c0 * DP + g * 8;
  FragBF A0a, A1a, B0a, B1a, A0b, A1b, B0b, B1b;

  auto load_frags = [&](FragBF& fA0, FragBF& fA1, FragBF& fB0, FragBF& fB1) {
    if constexpr (M1) {                              // tr <= 111 -> no clamp needed
      const u16b* xp0 = Xb + ltap * DP + lkc * 32;
      fA0.q[0] = *(const u32x4*)(xp0);
      fA0.q[1] = *(const u32x4*)(xp0 + 16);
      const u16b* xp1 = xp0 + 16 * DP;               // tr1 = tr0 + 16
      fA1.q[0] = *(const u32x4*)(xp1);
      fA1.q[1] = *(const u32x4*)(xp1 + 16);
    } else {                                         // bottom edge: clamp padded rows
      int xr0 = min(tr0 + ltap + (5 - K), REX - 1);
      const u16b* xp0 = Xe + xr0 * DP + lkc * 32 + g * 8;
      fA0.q[0] = *(const u32x4*)(xp0);
      fA0.q[1] = *(const u32x4*)(xp0 + 16);
    }
    const u16b* wp0 = Wp + (size_t)ltap * slabE + lkc * 32;
    fB0.q[0] = *(const u32x4*)(wp0);
    fB0.q[1] = *(const u32x4*)(wp0 + 16);
    if constexpr (N1) {
      const u16b* wp1 = wp0 + 16 * DP;               // c1 = c0 + 16
      fB1.q[0] = *(const u32x4*)(wp1);
      fB1.q[1] = *(const u32x4*)(wp1 + 16);
    }
    if (++ltap == K) { ltap = 0; ++lkc; }
  };
  auto do_wmmas = [&](const FragBF& fA0, const FragBF& fA1,
                      const FragBF& fB0, const FragBF& fB1) {
    acc00 = wmma_bf16(fA0, fB0, acc00);
    if constexpr (N1)       acc01 = wmma_bf16(fA0, fB1, acc01);
    if constexpr (M1)       acc10 = wmma_bf16(fA1, fB0, acc10);
    if constexpr (M1 && N1) acc11 = wmma_bf16(fA1, fB1, acc11);
  };

  // Branch-free modulo-2 software pipeline; tail handled at compile time.
  load_frags(A0a, A1a, B0a, B1a);
  constexpr int PAIRS = (itTot - 1) / 2;
  for (int p = 0; p < PAIRS; ++p) {
    load_frags(A0b, A1b, B0b, B1b);
    do_wmmas(A0a, A1a, B0a, B1a);
    load_frags(A0a, A1a, B0a, B1a);
    do_wmmas(A0b, A1b, B0b, B1b);
  }
  if constexpr ((itTot & 1) == 1) {
    do_wmmas(A0a, A1a, B0a, B1a);                    // odd tail (already loaded)
  } else {
    load_frags(A0b, A1b, B0b, B1b);                  // even tail
    do_wmmas(A0a, A1a, B0a, B1a);
    do_wmmas(A0b, A1b, B0b, B1b);
  }

  reduce_store(acc00, mt0, c0, g, rowsValid, biasrow, mxrow);
  if constexpr (N1)       reduce_store(acc01, mt0,     c0 + 16, g, rowsValid, biasrow, mxrow);
  if constexpr (M1)       reduce_store(acc10, mt0 + 1, c0,      g, rowsValid, biasrow, mxrow);
  if constexpr (M1 && N1) reduce_store(acc11, mt0 + 1, c0 + 16, g, rowsValid, biasrow, mxrow);
}

// -------- all blocks of the 9x7 tile grid for one kernel size K --------
template<int K>
__device__ __forceinline__ void conv_branch_K(
    const u16b* __restrict__ Xe, const u16b* __restrict__ Wb,
    unsigned* __restrict__ mxrow, const float* __restrict__ biasrow,
    int wv, int ln, int g) {
  for (int blk = wv; blk < 20; blk += 8) {           // wv is SGPR -> scalar loop/branches
    int mtb = blk >> 2, ntb = blk & 3;
    int mt0 = mtb * 2;
    int nt0 = ntb * 2;
    int tr0 = mt0 * 16 + ln;
    int c0  = nt0 * 16 + ln;
    bool m1v = (mtb < 4), n1v = (ntb < 3);           // scalar predicates
    if (m1v) {
      if (n1v) conv_block<K, true,  true >(Xe, Wb, mxrow, biasrow, tr0, c0, g, mt0);
      else     conv_block<K, true,  false>(Xe, Wb, mxrow, biasrow, tr0, c0, g, mt0);
    } else {
      if (n1v) conv_block<K, false, true >(Xe, Wb, mxrow, biasrow, tr0, c0, g, mt0);
      else     conv_block<K, false, false>(Xe, Wb, mxrow, biasrow, tr0, c0, g, mt0);
    }
  }
}

// -------- conv branches: tap-shifted bf16 WMMA + relu/bias/time-max --------
__global__ __launch_bounds__(256) void conv_kernel(const float* __restrict__ seq,
    const int* __restrict__ labels, const int* __restrict__ span,
    const u16b* __restrict__ Wr, const float* __restrict__ biasbuf,
    float* __restrict__ feat) {
  __shared__ u16b     Xe[REX * DP];    // gathered span, zero-padded rows/cols, bf16
  __shared__ unsigned mx[3 * CP];      // per-K column maxima (uint bits, values >= 0)

  int b = blockIdx.x, branch = blockIdx.y;
  int tid = threadIdx.x;
  int wv = __builtin_amdgcn_readfirstlane(tid >> 5); // scalar wave id
  int lane = tid & 31;
  int g = lane >> 4, ln = lane & 15;

  int s, e;
  if (branch == 0)      { s = span[b * 4 + 1]; e = span[b * 4 + 2]; }  // M: e1e..e2s
  else if (branch == 1) { s = span[b * 4 + 0]; e = span[b * 4 + 1]; }  // E1
  else                  { s = span[b * 4 + 2]; e = span[b * 4 + 3]; }  // E2

  const float* sq = seq + (size_t)b * TT * HH;
  const int*   lb = labels + (size_t)b * TT;
  const int spanLen = e - s;

  // ---- vectorized fill: rows 4..131, cols 0..511 (8 f32 -> 8 bf16 -> b128) ----
  for (int i = tid; i < TT * 64; i += 256) {
    int t = i >> 6, seg = i & 63;                    // col = seg*8
    int row = t + 4;
    Pack8 h;
    if (t < spanLen) {
      int pos = min(max(s + t, 0), TT - 1);
      const float* p = sq + (size_t)pos * HH + seg * 8;
      f32x4 x0 = *(const f32x4*)(p);
      f32x4 x1 = *(const f32x4*)(p + 4);
      h.s[0] = f2bf(x0.x); h.s[1] = f2bf(x0.y); h.s[2] = f2bf(x0.z); h.s[3] = f2bf(x0.w);
      h.s[4] = f2bf(x1.x); h.s[5] = f2bf(x1.y); h.s[6] = f2bf(x1.z); h.s[7] = f2bf(x1.w);
    } else {
      h.q = (u32x4){0u, 0u, 0u, 0u};
    }
    *(u32x4*)&Xe[(size_t)row * DP + seg * 8] = h.q;  // 16B-aligned (row stride 1088B)
  }
  // ---- zero padding rows (0..3, 132..135), all 68 segments ----
  for (int i = tid; i < 8 * 68; i += 256) {
    int pr = i / 68, seg = i - pr * 68;
    int row = (pr < 4) ? pr : (TT + pr);             // 0..3 and 132..135
    *(u32x4*)&Xe[(size_t)row * DP + seg * 8] = (u32x4){0u, 0u, 0u, 0u};
  }
  // ---- one-hot + pad cols 512..543 for data rows ----
  for (int i = tid; i < TT * 32; i += 256) {
    int t = i >> 5, col = HH + (i & 31);
    int row = t + 4;
    float v = 0.f;
    if (t < spanLen && col < DD) {
      int pos = min(max(s + t, 0), TT - 1);
      v = (lb[pos] == (col - HH)) ? 1.f : 0.f;
    }
    Xe[(size_t)row * DP + col] = f2bf(v);
  }
  for (int i = tid; i < 3 * CP; i += 256) mx[i] = 0u;
  __syncthreads();

  const size_t slab = (size_t)CP * DP;
  conv_branch_K<3>(Xe, Wr + (size_t)(branch * 12 + 0) * slab, &mx[0 * CP],
                   biasbuf + (branch * 3 + 0) * CP, wv, ln, g);
  conv_branch_K<4>(Xe, Wr + (size_t)(branch * 12 + 3) * slab, &mx[1 * CP],
                   biasbuf + (branch * 3 + 1) * CP, wv, ln, g);
  conv_branch_K<5>(Xe, Wr + (size_t)(branch * 12 + 7) * slab, &mx[2 * CP],
                   biasbuf + (branch * 3 + 2) * CP, wv, ln, g);

  __syncthreads();
  for (int i = tid; i < 300; i += 256) {
    int ki = i / 100, c = i - ki * 100;
    feat[(size_t)b * FEATP + branch * 300 + i] = __uint_as_float(mx[ki * CP + c]);
  }
}

// -------- final MLP: relu(feat @ W + b), bf16 WMMA --------
__global__ __launch_bounds__(256) void mlp_kernel(const float* __restrict__ feat,
    const u16b* __restrict__ Wm, const float* __restrict__ bias, float* __restrict__ out) {
  int tid = threadIdx.x;
  int wv = __builtin_amdgcn_readfirstlane(tid >> 5);
  int lane = tid & 31;
  int g = lane >> 4, ln = lane & 15;
  int mbase = blockIdx.x * 64, nbase = blockIdx.y * 64;
  for (int tt = wv * 2; tt < wv * 2 + 2; ++tt) {     // 16 tiles / 8 waves, scalar loop
    int mt = tt >> 2, nt = tt & 3;
    v8f acc = {};
    int arow = mbase + mt * 16 + ln;
    int bh   = nbase + nt * 16 + ln;
    const float* ap0  = feat + (size_t)arow * FEATP;
    const u16b*  wrow = Wm + (size_t)bh * FEATP;
    for (int kc = 0; kc < 65; ++kc) {
      FragBF fa, fb;
      const float* ap = ap0 + kc * 32 + g * 8;
      f32x4 x0 = *(const f32x4*)(ap);
      f32x4 x1 = *(const f32x4*)(ap + 4);
      f32x4 x2 = *(const f32x4*)(ap + 16);
      f32x4 x3 = *(const f32x4*)(ap + 20);
      fa.s[0]  = f2bf(x0.x); fa.s[1]  = f2bf(x0.y); fa.s[2]  = f2bf(x0.z); fa.s[3]  = f2bf(x0.w);
      fa.s[4]  = f2bf(x1.x); fa.s[5]  = f2bf(x1.y); fa.s[6]  = f2bf(x1.z); fa.s[7]  = f2bf(x1.w);
      fa.s[8]  = f2bf(x2.x); fa.s[9]  = f2bf(x2.y); fa.s[10] = f2bf(x2.z); fa.s[11] = f2bf(x2.w);
      fa.s[12] = f2bf(x3.x); fa.s[13] = f2bf(x3.y); fa.s[14] = f2bf(x3.z); fa.s[15] = f2bf(x3.w);
      const u16b* wp = wrow + kc * 32 + g * 8;
      fb.q[0] = *(const u32x4*)(wp);
      fb.q[1] = *(const u32x4*)(wp + 16);
      acc = wmma_bf16(fa, fb, acc);
    }
    float bv = bias[bh];
    #pragma unroll
    for (int r = 0; r < 8; ++r) {
      float y = acc[r] + bv;
      y = (y > 0.f) ? y : 0.f;
      out[(size_t)(mbase + mt * 16 + g * 8 + r) * HH + bh] = y;
    }
  }
}

// ---------------- host launcher ----------------
extern "C" void kernel_launch(void* const* d_in, const int* in_sizes, int n_in,
                              void* d_out, int out_size, void* d_ws, size_t ws_size,
                              hipStream_t stream) {
  (void)in_sizes; (void)n_in; (void)out_size; (void)ws_size;
  const float* seq    = (const float*)d_in[0];
  const int*   labels = (const int*)d_in[1];
  const int*   span   = (const int*)d_in[2];
  const int*   slens  = (const int*)d_in[3];
  ConvW cw;
  for (int br = 0; br < 3; ++br)
    for (int ki = 0; ki < 3; ++ki) {
      cw.w[br * 3 + ki] = (const float*)d_in[4 + br * 6 + ki * 2];
      cw.b[br * 3 + ki] = (const float*)d_in[4 + br * 6 + ki * 2 + 1];
    }
  const float* mlpw = (const float*)d_in[22];
  const float* mlpb = (const float*)d_in[23];
  float* out = (float*)d_out;

  char* ws = (char*)d_ws;
  u16b*  Wr      = (u16b*)(ws);                 // 36*112*544*2      = 4,386,816 B
  u16b*  Wm      = (u16b*)(ws + 4386816);       // 512*2080*2        = 2,129,920 B
  float* biasbuf = (float*)(ws + 6516736);      // 9*112*4           =     4,032 B
  float* feat    = (float*)(ws + 6520832);      // 1024*2080*4       = 8,519,680 B

  prep_conv_w_kernel<<<dim3(2048), dim3(256), 0, stream>>>(cw, Wr, biasbuf);
  prep_mlp_w_kernel <<<dim3(1024), dim3(256), 0, stream>>>(mlpw, Wm);
  fill_feat_kernel  <<<dim3(NB),   dim3(256), 0, stream>>>(seq, span, slens, feat);
  conv_kernel       <<<dim3(NB, 3), dim3(256), 0, stream>>>(seq, labels, span, Wr, biasbuf, feat);
  mlp_kernel        <<<dim3(16, 8), dim3(256), 0, stream>>>(feat, Wm, mlpb, out);
}
